// BiLSTM_GCN_1_45715631899085
// MI455X (gfx1250) — matrix operand order repaired and checked by reading, not compiled, not run on hardware
//
#include <hip/hip_runtime.h>
#include <hip/hip_bf16.h>
#include <math.h>

// ---------------- problem constants ----------------
#define GG    2048              // graphs (= LSTM batch)
#define TT    35                // nodes per graph (= seq len)
#define NN    (GG * TT)         // 71680 total nodes
#define DEGC  8
#define EE    (NN * DEGC)
#define INPUT_ 300
#define HLD   256               // hidden per direction
#define XPAD  320               // layer-0 input K padded to multiple of 32
#define K0TOT 576               // 320 + 256
#define K1TOT 768               // 512 + 256
#define MG_B  (GG / 16)         // 128  m-groups over batch rows
#define MG_N  (NN / 16)         // 4480 m-groups over node rows

typedef __attribute__((ext_vector_type(16))) __bf16 v16bf;
typedef __attribute__((ext_vector_type(8)))  float  v8f;

static __device__ __forceinline__ __bf16 f2bf(float f) {
  unsigned u = __float_as_uint(f);
  u = (u + 0x7FFFu + ((u >> 16) & 1u)) >> 16;           // round-to-nearest-even
  union { unsigned short s; __bf16 h; } c; c.s = (unsigned short)u;
  return c.h;
}
static __device__ __forceinline__ float sigf(float x) { return 1.f / (1.f + __expf(-x)); }

// A-fragment lane/element for (row m, k-within-32-tile kr):
//   lane = (m & 15) | (((kr>>3)&1) << 4) ;  elem j = (kr & 7) + (kr & 16 ? 8 : 0)
// Storage: [..., kt][mg][lane][16] -> 32 contiguous bytes per lane.
static __device__ __forceinline__ size_t afrag_off(int kr, int m) {
  const int ln = (m & 15) | (((kr >> 3) & 1) << 4);
  const int je = (kr & 7) + ((kr & 16) ? 8 : 0);
  return (size_t)ln * 16 + je;
}

// ---------------- conversion / packing kernels ----------------
// x [N,300] f32 -> A-fragment layout [t][kt=10][mg=128][512]
__global__ void k_convert_x(const float* __restrict__ x, __bf16* __restrict__ xb) {
  long long idx = (long long)blockIdx.x * 256 + threadIdx.x;
  if (idx >= (long long)NN * XPAD) return;
  int n = (int)(idx / XPAD);
  int c = (int)(idx % XPAD);
  int g = n / TT, t = n % TT;
  float v = (c < INPUT_) ? x[(long long)n * INPUT_ + c] : 0.f;
  const int kt = c >> 5, kr = c & 31;
  xb[(((size_t)t * (XPAD / 32) + kt) * MG_B + (g >> 4)) * 512 + afrag_off(kr, g)] = f2bf(v);
}

// concat weights -> B-fragment layout [dir][nkt][64 nt][32 lane][16]
//   B-frag: lane = (n&15) | (((k>>3)&1)<<4), elem = (k&7) + (k&16 ? 8 : 0)
__global__ void k_build_wcat(const float* __restrict__ Wih0, const float* __restrict__ Whh0,
                             const float* __restrict__ bih0, const float* __restrict__ bhh0,
                             const float* __restrict__ Wih1, const float* __restrict__ Whh1,
                             const float* __restrict__ bih1, const float* __restrict__ bhh1,
                             __bf16* __restrict__ wcat, float* __restrict__ biasc,
                             int din, int xpart, int Ktot) {
  long long idx = (long long)blockIdx.x * 256 + threadIdx.x;
  long long per = (long long)Ktot * 1024;
  if (idx >= 2 * per) return;
  int dir = (int)(idx / per);
  long long rem = idx - (long long)dir * per;
  int k = (int)(rem / 1024), j = (int)(rem % 1024);
  const float* Wih = dir ? Wih1 : Wih0;
  const float* Whh = dir ? Whh1 : Whh0;
  float v;
  if (k < din)        v = Wih[(long long)j * din + k];
  else if (k < xpart) v = 0.f;
  else                v = Whh[(long long)j * HLD + (k - xpart)];
  const int nkt = Ktot >> 5;
  wcat[((((size_t)dir * nkt + (k >> 5)) * 64 + (j >> 4)) * 32) * 16 + afrag_off(k & 31, j)] = f2bf(v);
  if (k == 0) {
    const float* bi = dir ? bih1 : bih0;
    const float* bh = dir ? bhh1 : bhh0;
    biasc[dir * 1024 + j] = bi[j] + bh[j];
  }
}

// gcn weight [512,256] -> B-fragment layout [kt=16][nt=16][32][16]
__global__ void k_build_wg(const float* __restrict__ w, __bf16* __restrict__ wg) {
  int i = blockIdx.x * 256 + threadIdx.x;
  if (i >= 512 * 256) return;
  int k = i >> 8, n = i & 255;
  wg[(((size_t)(k >> 5) * 16 + (n >> 4)) * 32) * 16 + afrag_off(k & 31, n)] = f2bf(w[i]);
}

// ---------------- fused LSTM step: barrier-free WMMA GEMM + gate activations ----------------
// grid (GG/32, 16, 2dirs), block 256 (8 waves = 2 m-tiles x 4 gates)
// XT = K-tiles of the x-part (10 for layer 0, 16 for layer 1); h-part is 8 tiles.
template <int XT>
__global__ __launch_bounds__(256)
void k_lstm_step(const __bf16* __restrict__ xfrag,              // [t][XT][MG_B][512]
                 const __bf16* __restrict__ hfrag,              // [dir][8][MG_B][512]
                 __bf16* __restrict__ hfrag_out,
                 float* __restrict__ cstate,                    // [dir][B][256] f32
                 const __bf16* __restrict__ wfrag,              // [dir][XT+8][64][512]
                 const float* __restrict__ biasc,               // [dir][1024]
                 __bf16* __restrict__ lout_frag,                // layer0: [t][16][MG_B][512] (col = dir*256+j)
                 __bf16* __restrict__ lout_node,                // layer1: [16][MG_N][512], row n = g*35+t
                 int step) {
  constexpr int NKT = XT + 8;
  __shared__ float Gs[2 * 4 * 16 * 16];

  const int tid  = threadIdx.x;
  const int dir  = blockIdx.z;
  const int t    = (dir == 0) ? step : (TT - 1 - step);
  const int mb   = blockIdx.x * 32;
  const int jb   = blockIdx.y;            // 16-col group of j
  const int lane = tid & 31;
  const int wave = tid >> 5;
  const int mtile = wave & 1;
  const int gate  = wave >> 1;
  const int mg = (mb >> 4) + mtile;
  const int nt = gate * 16 + jb;

  const __bf16* wb = wfrag + (size_t)dir * NKT * 64 * 512;
  const __bf16* hb = hfrag + (size_t)dir * (8 * MG_B * 512);

  v8f acc = {0.f, 0.f, 0.f, 0.f, 0.f, 0.f, 0.f, 0.f};
  // phase 1: x-part, software-pipelined one K-tile ahead
  {
    const __bf16* ap = xfrag + ((size_t)t * XT * MG_B + mg) * 512 + lane * 16;
    const __bf16* bp = wb + ((size_t)nt * 32 + lane) * 16;
    v16bf a = *(const v16bf*)ap;
    v16bf b = *(const v16bf*)bp;
#pragma unroll
    for (int kk = 1; kk < XT; ++kk) {
      const v16bf a2 = *(const v16bf*)(ap + (size_t)kk * (MG_B * 512));
      const v16bf b2 = *(const v16bf*)(bp + (size_t)kk * (64 * 512));
      acc = __builtin_amdgcn_wmma_f32_16x16x32_bf16(false, a, false, b, (short)0, acc, false, false);
      a = a2; b = b2;
    }
    acc = __builtin_amdgcn_wmma_f32_16x16x32_bf16(false, a, false, b, (short)0, acc, false, false);
  }
  // phase 2: recurrent h-part (8 K-tiles), same pipelining
  {
    const __bf16* ap = hb + (size_t)mg * 512 + lane * 16;
    const __bf16* bp = wb + (((size_t)XT * 64 + nt) * 32 + lane) * 16;
    v16bf a = *(const v16bf*)ap;
    v16bf b = *(const v16bf*)bp;
#pragma unroll
    for (int kk = 1; kk < 8; ++kk) {
      const v16bf a2 = *(const v16bf*)(ap + (size_t)kk * (MG_B * 512));
      const v16bf b2 = *(const v16bf*)(bp + (size_t)kk * (64 * 512));
      acc = __builtin_amdgcn_wmma_f32_16x16x32_bf16(false, a, false, b, (short)0, acc, false, false);
      a = a2; b = b2;
    }
    acc = __builtin_amdgcn_wmma_f32_16x16x32_bf16(false, a, false, b, (short)0, acc, false, false);
  }
  {   // exchange per-gate accumulators through LDS: [mtile][gate][16][16]
    const int nloc = lane & 15;
    const int mhi  = (lane >> 4) ? 8 : 0;
#pragma unroll
    for (int r = 0; r < 8; ++r)
      Gs[((mtile * 4 + gate) * 16 + (r + mhi)) * 16 + nloc] = acc[r];
  }
  __syncthreads();
  // fused i,f,g,o activations: 2 cells per thread
  const float* bb = biasc + dir * 1024;
  float*  cs   = cstate    + (size_t)dir * GG * HLD;
  __bf16* hofr = hfrag_out + (size_t)dir * (8 * MG_B * 512);
#pragma unroll
  for (int it = 0; it < 2; ++it) {
    const int q = tid + it * 256;
    const int mloc = q >> 4;
    const int n = q & 15;
    const int mt = mloc >> 4, mr = mloc & 15;
    const int bg = mb + mloc;               // batch row (= graph id)
    const int jg = jb * 16 + n;             // hidden index 0..255
    const float iv = Gs[((mt * 4 + 0) * 16 + mr) * 16 + n] + bb[jg];
    const float fv = Gs[((mt * 4 + 1) * 16 + mr) * 16 + n] + bb[256 + jg];
    const float gv = Gs[((mt * 4 + 2) * 16 + mr) * 16 + n] + bb[512 + jg];
    const float ov = Gs[((mt * 4 + 3) * 16 + mr) * 16 + n] + bb[768 + jg];
    const size_t ci = (size_t)bg * HLD + jg;
    const float cp = cs[ci];
    const float cn = sigf(fv) * cp + sigf(iv) * tanhf(gv);
    const float hv = sigf(ov) * tanhf(cn);
    cs[ci] = cn;
    const __bf16 hbf = f2bf(hv);
    // recurrent state in A-fragment layout
    hofr[((size_t)(jg >> 5) * MG_B + (bg >> 4)) * 512 + afrag_off(jg & 31, bg)] = hbf;
    if (lout_frag) {      // layer-0 output, consumed as layer-1 A-matrix (batch rows)
      const int col = dir * 256 + jg;
      lout_frag[(((size_t)t * 16 + (col >> 5)) * MG_B + (bg >> 4)) * 512 + afrag_off(col & 31, bg)] = hbf;
    }
    if (lout_node) {      // layer-1 output, consumed as GCN A-matrix (node rows)
      const int nid = bg * TT + t;
      const int col = dir * 256 + jg;
      lout_node[((size_t)(col >> 5) * MG_N + (nid >> 4)) * 512 + afrag_off(col & 31, nid)] = hbf;
    }
  }
}

// ---------------- GCN GEMM: gh[N,256] = o @ W scaled by rsqrt(out_deg), barrier-free -------
// grid (NN/32, 4), block 256 (8 waves = 2 m-tiles x 4 n-tiles)
__global__ __launch_bounds__(256)
void k_gcn_gemm(const __bf16* __restrict__ h1,  // node A-fragments [16][MG_N][512]
                const __bf16* __restrict__ wg,  // B-fragments [16][16][32][16]
                const float* __restrict__ rso, float* __restrict__ gh) {
  const int tid  = threadIdx.x;
  const int lane = tid & 31;
  const int wave = tid >> 5;
  const int mtile = wave & 1, ntile = wave >> 1;
  const int mg = blockIdx.x * 2 + mtile;
  const int nt = blockIdx.y * 4 + ntile;

  const __bf16* ap = h1 + (size_t)mg * 512 + lane * 16;
  const __bf16* bp = wg + ((size_t)nt * 32 + lane) * 16;
  v8f acc = {0.f, 0.f, 0.f, 0.f, 0.f, 0.f, 0.f, 0.f};
  v16bf a = *(const v16bf*)ap;
  v16bf b = *(const v16bf*)bp;
#pragma unroll
  for (int kk = 1; kk < 16; ++kk) {
    const v16bf a2 = *(const v16bf*)(ap + (size_t)kk * (MG_N * 512));
    const v16bf b2 = *(const v16bf*)(bp + (size_t)kk * (16 * 512));
    acc = __builtin_amdgcn_wmma_f32_16x16x32_bf16(false, a, false, b, (short)0, acc, false, false);
    a = a2; b = b2;
  }
  acc = __builtin_amdgcn_wmma_f32_16x16x32_bf16(false, a, false, b, (short)0, acc, false, false);

  const int nloc = lane & 15;
  const int mhi  = (lane >> 4) ? 8 : 0;
#pragma unroll
  for (int r = 0; r < 8; ++r) {
    const int m = mg * 16 + r + mhi;
    gh[(size_t)m * 256 + nt * 16 + nloc] = acc[r] * rso[m];
  }
}

// ---------------- graph plumbing ----------------
__global__ void k_count_deg(const int* __restrict__ src, const int* __restrict__ dst,
                            float* __restrict__ oc, float* __restrict__ ic) {
  int e = blockIdx.x * 256 + threadIdx.x;
  if (e >= EE) return;
  atomicAdd(&oc[src[e]], 1.f);
  atomicAdd(&ic[dst[e]], 1.f);
}
__global__ void k_fin_deg(float* __restrict__ oc, float* __restrict__ ic) {
  int n = blockIdx.x * 256 + threadIdx.x;
  if (n >= NN) return;
  oc[n] = rsqrtf(fmaxf(oc[n], 1.f));
  ic[n] = rsqrtf(fmaxf(ic[n], 1.f));
}
// dst = repeat(arange(N), 8) in the reference => aggregation is a per-node gather (deterministic)
__global__ void k_gcn_gather(const float* __restrict__ gh, const int* __restrict__ src,
                             const float* __restrict__ rsi, const float* __restrict__ bias,
                             float* __restrict__ agg) {
  const int n = blockIdx.x;
  const int j = threadIdx.x;
  float acc = 0.f;
#pragma unroll
  for (int d = 0; d < DEGC; ++d) {
    const int s = src[n * DEGC + d];
    acc += gh[(size_t)s * 256 + j];
  }
  const float v = acc * rsi[n] + bias[j];
  agg[(size_t)n * 256 + j] = v > 0.f ? v : 0.f;
}
__global__ void k_graph_mean(const float* __restrict__ agg, float* __restrict__ rep) {
  const int g = blockIdx.x;
  const int j = threadIdx.x;
  float acc = 0.f;
  for (int t = 0; t < TT; ++t) acc += agg[((size_t)g * TT + t) * 256 + j];
  rep[(size_t)g * 256 + j] = acc * (1.f / (float)TT);
}
__global__ void k_predict(const float* __restrict__ r1, const float* __restrict__ r2,
                          const float* __restrict__ pw, const float* __restrict__ pb,
                          float* __restrict__ out) {
  int g = blockIdx.x * 256 + threadIdx.x;
  if (g >= GG) return;
  float a0 = 0.f, a1 = 0.f;
  for (int j = 0; j < 256; ++j) {
    const float d = tanhf(r1[(size_t)g * 256 + j] - r2[(size_t)g * 256 + j]);
    a0 += d * pw[j * 2 + 0];
    a1 += d * pw[j * 2 + 1];
  }
  out[g * 2 + 0] = a0 + pb[0];
  out[g * 2 + 1] = a1 + pb[1];
}

// ---------------- host launch ----------------
extern "C" void kernel_launch(void* const* d_in, const int* in_sizes, int n_in,
                              void* d_out, int out_size, void* d_ws, size_t ws_size,
                              hipStream_t stream) {
  (void)in_sizes; (void)n_in; (void)out_size; (void)ws_size;

  char* ws = (char*)d_ws;
  size_t off = 0;
  auto take = [&](size_t bytes) -> void* {
    void* p = ws + off;
    off = (off + bytes + 255) & ~(size_t)255;
    return p;
  };
  const size_t HSTHALF = (size_t)2 * GG * HLD;                   // elems per ping-pong half
  __bf16* XB  = (__bf16*)take((size_t)TT * GG * XPAD * 2);       // x A-frags [t][10][128][512]
  __bf16* H0  = (__bf16*)take((size_t)TT * GG * 512 * 2);        // layer0 out A-frags [t][16][128][512]
  __bf16* H1  = (__bf16*)take((size_t)NN * 512 * 2);             // layer1 out A-frags [16][4480][512]
  __bf16* HST = (__bf16*)take((size_t)2 * HSTHALF * 2);          // h ping-pong frags [pp][dir][8][128][512]
  float*  CST = (float*) take((size_t)2 * GG * HLD * 4);         // c state [dir][B][256]
  __bf16* WC  = (__bf16*)take((size_t)2 * K1TOT * 1024 * 2);     // concat weight B-frags
  float*  BC  = (float*) take((size_t)2 * 1024 * 4);             // combined bias
  __bf16* WG  = (__bf16*)take((size_t)512 * 256 * 2);            // gcn weight B-frags
  float*  GH  = (float*) take((size_t)NN * 256 * 4);             // o@W * rsqrt(outdeg)
  float*  AGG = (float*) take((size_t)NN * 256 * 4);             // relu'd node features
  float*  RSO = (float*) take((size_t)NN * 4);
  float*  RSI = (float*) take((size_t)NN * 4);
  float*  REP0 = (float*)take((size_t)GG * 256 * 4);
  float*  REP1 = (float*)take((size_t)GG * 256 * 4);
  float*  REP[2] = {REP0, REP1};

  for (int br = 0; br < 2; ++br) {
    const float* xin = (const float*)d_in[br];
    const int* src = (const int*)d_in[2 + br * 2];
    const int* dst = (const int*)d_in[3 + br * 2];
    const int lb = 6 + br * 16;

    k_convert_x<<<(NN * XPAD + 255) / 256, 256, 0, stream>>>(xin, XB);

    for (int L = 0; L < 2; ++L) {
      const int din   = L ? 512 : INPUT_;
      const int xpart = L ? 512 : XPAD;
      const int Ktot  = L ? K1TOT : K0TOT;
      const float* Wih0 = (const float*)d_in[lb + L * 8 + 0];
      const float* Whh0 = (const float*)d_in[lb + L * 8 + 1];
      const float* bih0 = (const float*)d_in[lb + L * 8 + 2];
      const float* bhh0 = (const float*)d_in[lb + L * 8 + 3];
      const float* Wih1 = (const float*)d_in[lb + L * 8 + 4];
      const float* Whh1 = (const float*)d_in[lb + L * 8 + 5];
      const float* bih1 = (const float*)d_in[lb + L * 8 + 6];
      const float* bhh1 = (const float*)d_in[lb + L * 8 + 7];
      {
        const long long tot = 2LL * Ktot * 1024;
        k_build_wcat<<<(int)((tot + 255) / 256), 256, 0, stream>>>(
            Wih0, Whh0, bih0, bhh0, Wih1, Whh1, bih1, bhh1, WC, BC, din, xpart, Ktot);
      }
      hipMemsetAsync(HST, 0, HSTHALF * 2, stream);               // zero ping h
      hipMemsetAsync(CST, 0, (size_t)2 * GG * HLD * 4, stream);  // zero c

      int pp = 0;
      for (int s = 0; s < TT; ++s) {
        __bf16* hin  = HST + (size_t)pp       * HSTHALF;
        __bf16* hout = HST + (size_t)(1 - pp) * HSTHALF;
        if (L == 0) {
          k_lstm_step<10><<<dim3(GG / 32, 16, 2), 256, 0, stream>>>(
              XB, hin, hout, CST, WC, BC, H0, (__bf16*)nullptr, s);
        } else {
          k_lstm_step<16><<<dim3(GG / 32, 16, 2), 256, 0, stream>>>(
              H0, hin, hout, CST, WC, BC, (__bf16*)nullptr, H1, s);
        }
        pp ^= 1;
      }
    }

    // GCN
    k_build_wg<<<(512 * 256 + 255) / 256, 256, 0, stream>>>((const float*)d_in[38 + br * 2], WG);
    hipMemsetAsync(RSO, 0, (size_t)NN * 4, stream);
    hipMemsetAsync(RSI, 0, (size_t)NN * 4, stream);
    k_count_deg<<<(EE + 255) / 256, 256, 0, stream>>>(src, dst, RSO, RSI);
    k_fin_deg<<<(NN + 255) / 256, 256, 0, stream>>>(RSO, RSI);
    k_gcn_gemm<<<dim3(NN / 32, 4), 256, 0, stream>>>(H1, WG, RSO, GH);
    k_gcn_gather<<<NN, 256, 0, stream>>>(GH, src, RSI, (const float*)d_in[39 + br * 2], AGG);
    k_graph_mean<<<GG, 256, 0, stream>>>(AGG, REP[br]);
  }

  k_predict<<<GG / 256, 256, 0, stream>>>(
      REP[0], REP[1], (const float*)d_in[42], (const float*)d_in[43], (float*)d_out);
}
